// ConjunctionLayer_52493090291974
// MI455X (gfx1250) — 8-bit matrix operand order repaired and checked by reading, compiled, not run on hardware
//
#include <hip/hip_runtime.h>

typedef __attribute__((ext_vector_type(16))) int   v16i;
typedef __attribute__((ext_vector_type(8)))  float v8f;

#define BATCH   8192
#define IN_DIM  4096
#define N_RULES 2048
#define KB      (IN_DIM / 128)
#define FP8_ONE 0x38u

__global__ void __launch_bounds__(256) pack_a_fp8(const float* __restrict__ x,
                                                  unsigned* __restrict__ apack) {
  unsigned tid    = blockIdx.x * 256u + threadIdx.x;
  unsigned frag   = tid >> 9;
  unsigned within = tid & 511u;
  unsigned lane   = within >> 4;
  unsigned v      = within & 15u;
  unsigned mt     = frag >> 5;
  unsigned kb     = frag & 31u;
  unsigned m      = mt * 16u + (lane & 15u);
  unsigned half   = lane >> 4;
  unsigned sub    = v >> 3;
  unsigned v8     = v & 7u;
  unsigned g      = v8 >> 1;
  unsigned k0     = kb * 128u + sub * 64u + g * 16u + half * 8u + (v8 & 1u) * 4u;

  const float* row = x + (size_t)m * IN_DIM + k0;
  unsigned d = 0;
#pragma unroll
  for (int b = 0; b < 4; ++b) {
    float xv = row[b];
    d |= ((xv < 1.0f) ? FP8_ONE : 0u) << (8 * b);
  }
  apack[tid] = d;
}

__global__ void __launch_bounds__(256) pack_b_fp8(const float* __restrict__ W,
                                                  unsigned* __restrict__ bpack) {
  unsigned tid    = blockIdx.x * 256u + threadIdx.x;
  unsigned frag   = tid >> 9;
  unsigned within = tid & 511u;
  unsigned lane   = within >> 4;
  unsigned v      = within & 15u;
  unsigned nt     = frag >> 5;
  unsigned kb     = frag & 31u;
  unsigned n      = nt * 16u + (lane & 15u);
  unsigned half   = lane >> 4;
  unsigned q      = v >> 2;
  unsigned k0     = kb * 128u + (v >> 3) * 64u + ((v & 7u) >> 1) * 16u + half * 8u + (v & 1u) * 4u;

  unsigned d = 0;
#pragma unroll
  for (int b = 0; b < 4; ++b) {
    float wv = W[(size_t)(k0 + b) * N_RULES + n];
    d |= ((wv > 0.5f) ? FP8_ONE : 0u) << (8 * b);
  }
  bpack[tid] = d;
}

__global__ void __launch_bounds__(256, 1)
conj_gemm_fp8(const v16i* __restrict__ apack, const v16i* __restrict__ bpack,
              float* __restrict__ out) {
  const unsigned lane   = threadIdx.x & 31u;
  const unsigned warp   = threadIdx.x >> 5;
  const unsigned wt     = blockIdx.x * 8u + warp;
  const unsigned m_tile = wt >> 5;
  const unsigned n_tile = wt & 31u;

  const size_t SUB = (size_t)KB * 32u;

  const v16i* ap = apack + (size_t)(m_tile * 4u) * SUB + lane;
  const v16i* bp = bpack + (size_t)(n_tile * 4u) * SUB + lane;

  v8f acc[4][4];
#pragma unroll
  for (int mi = 0; mi < 4; ++mi)
#pragma unroll
    for (int ni = 0; ni < 4; ++ni) acc[mi][ni] = v8f{};

  v16i a[4];
  v16i bcur = bp[0];

  for (unsigned kb = 0; kb < KB; ++kb) {
    const size_t ko   = (size_t)kb * 32u;
    const size_t konx = (size_t)((kb + 1u < KB) ? kb + 1u : kb) * 32u;

    __builtin_prefetch(ap + ko + 64u, 0, 1);
    __builtin_prefetch(bp + ko + 64u, 0, 1);

#pragma unroll
    for (int mi = 0; mi < 4; ++mi) a[mi] = ap[ko + (size_t)mi * SUB];

#pragma unroll
    for (int ni = 0; ni < 4; ++ni) {
      const size_t nidx = (ni < 3) ? (ko + (size_t)(ni + 1) * SUB) : konx;
      v16i bnxt = bp[nidx];

      acc[0][ni] = __builtin_amdgcn_wmma_f32_16x16x128_fp8_fp8(
          a[0], bcur, (short)0, acc[0][ni], false, false);
      acc[1][ni] = __builtin_amdgcn_wmma_f32_16x16x128_fp8_fp8(
          a[1], bcur, (short)0, acc[1][ni], false, false);
      acc[2][ni] = __builtin_amdgcn_wmma_f32_16x16x128_fp8_fp8(
          a[2], bcur, (short)0, acc[2][ni], false, false);
      acc[3][ni] = __builtin_amdgcn_wmma_f32_16x16x128_fp8_fp8(
          a[3], bcur, (short)0, acc[3][ni], false, false);

      bcur = bnxt;
    }
  }

  const unsigned ncol = n_tile * 64u + (lane & 15u);
  const unsigned mhi  = (lane >> 4) << 3;
#pragma unroll
  for (int mi = 0; mi < 4; ++mi) {
#pragma unroll
    for (int ni = 0; ni < 4; ++ni) {
#pragma unroll
      for (int r = 0; r < 8; ++r) {
        unsigned m = m_tile * 64u + (unsigned)mi * 16u + (unsigned)r + mhi;
        unsigned n = ncol + (unsigned)ni * 16u;
        out[(size_t)m * N_RULES + n] = (acc[mi][ni][r] > 0.0f) ? 0.0f : 1.0f;
      }
    }
  }
}

extern "C" void kernel_launch(void* const* d_in, const int* in_sizes, int n_in,
                              void* d_out, int out_size, void* d_ws, size_t ws_size,
                              hipStream_t stream) {
  const float* x = (const float*)d_in[0];
  const float* W = (const float*)d_in[1];
  float* out = (float*)d_out;

  unsigned* apack = (unsigned*)d_ws;
  unsigned* bpack = (unsigned*)((char*)d_ws + (size_t)BATCH * IN_DIM);

  pack_a_fp8<<<(BATCH * IN_DIM / 4) / 256, 256, 0, stream>>>(x, apack);
  pack_b_fp8<<<(IN_DIM * N_RULES / 4) / 256, 256, 0, stream>>>(W, bpack);

  conj_gemm_fp8<<<512, 256, 0, stream>>>((const v16i*)apack,
                                         (const v16i*)bpack, out);
}
